// ScaledDotProductAttention_20547123544783
// MI455X (gfx1250) — matrix operand order repaired and checked by reading, not compiled, not run on hardware
//
#include <hip/hip_runtime.h>
#include <hip/hip_bf16.h>
#include <math.h>

typedef __attribute__((ext_vector_type(16))) __bf16 v16bf;
typedef __attribute__((ext_vector_type(8)))  __bf16 v8bf;
typedef __attribute__((ext_vector_type(8)))  float  v8f;

#define S_LEN     2048
#define D_DIM     64
#define INV_SCALE 0.125f      // 1/sqrt(64)
#define MASK_FILL -10000.0f

#define KV_STRIDE 68          // f32/row of raw DMA tile: 64 + 4 pad (272B rows, 16B aligned)
#define KB_STRIDE 72          // bf16/row of packed K tile (144B rows, 16B aligned)
#define VT_STRIDE 40          // bf16/row of transposed V tile (80B rows, 16B aligned)
#define P_STRIDE  40          // bf16/row of P staging (80B rows, 16B aligned)
#define M_STRIDE  48          // bytes/row of mask tile (16B aligned)

// ---- CDNA5 async DMA: global -> LDS, 16B per lane, tracked by ASYNCcnt ----
__device__ __forceinline__ void async_cp16(unsigned int lds_addr, const void* gaddr) {
    asm volatile("global_load_async_to_lds_b128 %0, %1, off"
                 :: "v"(lds_addr), "v"(gaddr)
                 : "memory");
}
__device__ __forceinline__ void wait_async0() {
    asm volatile("s_wait_asynccnt 0x0" ::: "memory");
}
__device__ __forceinline__ unsigned int ldsa(const void* p) {
    return (unsigned int)(size_t)p;          // low 32 bits of flat shared addr = LDS offset
}
// pack high-16 (bf16-truncate) of two f32 bit patterns into one dword: one v_perm_b32
__device__ __forceinline__ unsigned int pack2(unsigned int f0, unsigned int f1) {
    return __builtin_amdgcn_perm(f1, f0, 0x07060302u);
}
__device__ __forceinline__ __bf16 bf_hi(float f) {   // f32 -> bf16 by truncation
    union { unsigned short u; __bf16 b; } cv;
    cv.u = (unsigned short)(__float_as_uint(f) >> 16);
    return cv.b;
}

__global__ __launch_bounds__(128)
void sdpa_fused_kernel(const float* __restrict__ Q,
                       const float* __restrict__ K,
                       const float* __restrict__ V,
                       const unsigned char* __restrict__ M,
                       float* __restrict__ ctx,
                       float* __restrict__ prob)
{
    __shared__ float          Kf[2][32][KV_STRIDE];   // raw f32 K chunk (async DMA dst)
    __shared__ float          Vf[2][32][KV_STRIDE];   // raw f32 V chunk
    __shared__ unsigned char  Mb[2][64][M_STRIDE];    // mask tile for whole block
    __shared__ unsigned short Kb16[32][KB_STRIDE];    // packed bf16 K (keys x dims)
    __shared__ unsigned short Vt16[D_DIM][VT_STRIDE]; // packed bf16 V^T (dims x keys)
    __shared__ unsigned short Pb[4][16][P_STRIDE];    // per-wave P tile (bf16 bits)

    const int tid    = threadIdx.x;
    const int wave   = tid >> 5;
    const int lane   = tid & 31;
    const int r      = lane & 15;
    const int halfId = lane >> 4;

    const int bh    = blockIdx.y;
    const int qbase = blockIdx.x * 64;
    const int q0    = qbase + wave * 16;

    const float*         Qp = Q + (size_t)bh * S_LEN * D_DIM;
    const float*         Kp = K + (size_t)bh * S_LEN * D_DIM;
    const float*         Vp = V + (size_t)bh * S_LEN * D_DIM;
    const unsigned char* Mp = M + (size_t)bh * S_LEN * S_LEN;
    float*               Cp = ctx  + (size_t)bh * S_LEN * D_DIM;
    float*               Pp = prob + (size_t)bh * S_LEN * S_LEN;

    // ---- async-issue helpers (32-key chunk each) ----
    auto issueK = [&](int kc, int b) {
        const float* gk = Kp + (size_t)kc * D_DIM;
        #pragma unroll
        for (int j = 0; j < 4; ++j) {                 // 512 x 16B segs / 128 thr
            int i = tid + j * 128;
            int kr = i >> 4, kq = (i & 15) << 2;
            async_cp16(ldsa(&Kf[b][kr][kq]), gk + (size_t)i * 4);
        }
    };
    auto issueV = [&](int kc, int b) {
        const float* gv = Vp + (size_t)kc * D_DIM;
        #pragma unroll
        for (int j = 0; j < 4; ++j) {
            int i = tid + j * 128;
            int kr = i >> 4, kq = (i & 15) << 2;
            async_cp16(ldsa(&Vf[b][kr][kq]), gv + (size_t)i * 4);
        }
    };
    auto issueM = [&](int kc, int b) {                // 64 rows x 32B = 128 x 16B segs
        int gr = tid >> 1, gc = (tid & 1) * 16;
        async_cp16(ldsa(&Mb[b][gr][gc]), Mp + (size_t)(qbase + gr) * S_LEN + kc + gc);
    };

    // ---- convert raw f32 -> packed bf16 tiles (once per chunk, shared by 4 waves) ----
    auto convK = [&](int fb) {                        // row-major: v_perm pair packing
        int row = tid >> 2, d0 = (tid & 3) * 16;
        const uint4* s = (const uint4*)&Kf[fb][row][d0];
        uint4 a = s[0], b = s[1], c = s[2], d = s[3];
        uint4 o0, o1;
        o0.x = pack2(a.x, a.y); o0.y = pack2(a.z, a.w);
        o0.z = pack2(b.x, b.y); o0.w = pack2(b.z, b.w);
        o1.x = pack2(c.x, c.y); o1.y = pack2(c.z, c.w);
        o1.z = pack2(d.x, d.y); o1.w = pack2(d.z, d.w);
        *(uint4*)&Kb16[row][d0]     = o0;
        *(uint4*)&Kb16[row][d0 + 8] = o1;
    };
    auto convV = [&](int fb) {                        // transpose: strided hi-u16 gather, once
        const unsigned short* vs = (const unsigned short*)&Vf[fb][0][0];
        #pragma unroll
        for (int j = 0; j < 2; ++j) {
            int s = tid + j * 128;                    // 256 segments of 8 keys
            int d = s >> 2, k0 = (s & 3) * 8;
            unsigned int w[4];
            #pragma unroll
            for (int q = 0; q < 4; ++q) {
                unsigned int lo = vs[((k0 + 2*q    ) * KV_STRIDE + d) * 2 + 1];
                unsigned int hi = vs[((k0 + 2*q + 1) * KV_STRIDE + d) * 2 + 1];
                w[q] = lo | (hi << 16);
            }
            uint4 o; o.x = w[0]; o.y = w[1]; o.z = w[2]; o.w = w[3];
            *(uint4*)&Vt16[d][k0] = o;
        }
    };

    // ---- fragment loaders: 2 x ds_load_b128 each ----
    auto loadA = [&](const unsigned short* rowp, int col0) -> v16bf {
        // A-operand 16-bit pattern: e0-7 -> k {col0+8h .. +7}, e8-15 -> k {col0+16+8h .. +7}
        v8bf lo = *(const v8bf*)(rowp + col0 + 8 * halfId);
        v8bf hi = *(const v8bf*)(rowp + col0 + 16 + 8 * halfId);
        return __builtin_shufflevector(lo, hi, 0,1,2,3,4,5,6,7,8,9,10,11,12,13,14,15);
    };

    // ---- Q tile -> two bf16 A fragments (one-time, from global) ----
    v16bf aQ0, aQ1;
    #pragma unroll
    for (int e = 0; e < 16; ++e) {
        int kd = e + ((e < 8) ? 0 : 8) + 8 * halfId;
        aQ0[e] = bf_hi(Qp[(q0 + r) * D_DIM + kd]);
        aQ1[e] = bf_hi(Qp[(q0 + r) * D_DIM + 32 + kd]);
    }

    float pm[8], pl[8];
    #pragma unroll
    for (int i = 0; i < 8; ++i) { pm[i] = -3.0e38f; pl[i] = 0.0f; }

    // ================= pass 1: online row max / sum(exp) =================
    int fb = 0;
    issueK(0, fb); issueM(0, fb);
    wait_async0();
    __syncthreads();

    for (int kc = 0; kc < S_LEN; kc += 32) {
        if (kc + 32 < S_LEN) { issueK(kc + 32, fb ^ 1); issueM(kc + 32, fb ^ 1); }
        convK(fb);
        __syncthreads();                  // Kb16 ready for all waves

        #pragma unroll
        for (int t = 0; t < 2; ++t) {
            const unsigned short* krow = &Kb16[t * 16 + r][0];  // B = K^T: A-pattern row read
            v16bf b0 = loadA(krow, 0);
            v16bf b1 = loadA(krow, 32);
            v8f c = {0.f,0.f,0.f,0.f,0.f,0.f,0.f,0.f};
            c = __builtin_amdgcn_wmma_f32_16x16x32_bf16(false, aQ0, false, b0, (short)0, c, false, false);
            c = __builtin_amdgcn_wmma_f32_16x16x32_bf16(false, aQ1, false, b1, (short)0, c, false, false);

            #pragma unroll
            for (int i = 0; i < 8; ++i) {             // C layout: M=i+8*halfId, N=r
                int mrow = i + 8 * halfId;
                float s = Mb[fb][wave * 16 + mrow][t * 16 + r] ? MASK_FILL : c[i] * INV_SCALE;
                float mn = fmaxf(pm[i], s);
                pl[i] = pl[i] * __expf(pm[i] - mn) + __expf(s - mn);
                pm[i] = mn;
            }
        }
        wait_async0();                    // next chunk's DMA has overlapped the work above
        __syncthreads();
        fb ^= 1;
    }

    // merge stats across the 16-lane half-wave (halves hold different M rows)
    #pragma unroll
    for (int off = 1; off < 16; off <<= 1) {
        #pragma unroll
        for (int i = 0; i < 8; ++i) {
            float mo = __shfl_xor(pm[i], off, 32);
            float lo = __shfl_xor(pl[i], off, 32);
            float mn = fmaxf(pm[i], mo);
            pl[i] = pl[i] * __expf(pm[i] - mn) + lo * __expf(mo - mn);
            pm[i] = mn;
        }
    }
    float rl[8];
    #pragma unroll
    for (int i = 0; i < 8; ++i) rl[i] = 1.0f / pl[i];

    // ================= pass 2: stream probs (NT) + accumulate P·V =================
    v8f acc[4];
    #pragma unroll
    for (int nt = 0; nt < 4; ++nt)
        acc[nt] = (v8f){0.f,0.f,0.f,0.f,0.f,0.f,0.f,0.f};

    issueK(0, fb); issueV(0, fb); issueM(0, fb);
    wait_async0();
    __syncthreads();

    for (int kc = 0; kc < S_LEN; kc += 32) {
        if (kc + 32 < S_LEN) {
            issueK(kc + 32, fb ^ 1); issueV(kc + 32, fb ^ 1); issueM(kc + 32, fb ^ 1);
        }
        convK(fb); convV(fb);
        __syncthreads();                  // Kb16 / Vt16 ready

        #pragma unroll
        for (int t = 0; t < 2; ++t) {
            const unsigned short* krow = &Kb16[t * 16 + r][0];
            v16bf b0 = loadA(krow, 0);
            v16bf b1 = loadA(krow, 32);
            v8f c = {0.f,0.f,0.f,0.f,0.f,0.f,0.f,0.f};
            c = __builtin_amdgcn_wmma_f32_16x16x32_bf16(false, aQ0, false, b0, (short)0, c, false, false);
            c = __builtin_amdgcn_wmma_f32_16x16x32_bf16(false, aQ1, false, b1, (short)0, c, false, false);

            const int kcol = kc + t * 16 + r;
            #pragma unroll
            for (int i = 0; i < 8; ++i) {
                int mrow = i + 8 * halfId;
                int qrow = q0 + mrow;
                float s = Mb[fb][wave * 16 + mrow][t * 16 + r] ? MASK_FILL : c[i] * INV_SCALE;
                float p = __expf(s - pm[i]) * rl[i];
                // 537 MB write-once stream: non-temporal keeps L2 for K/V/mask reuse
                __builtin_nontemporal_store(p, &Pp[(size_t)qrow * S_LEN + kcol]);
                Pb[wave][mrow][t * 16 + r] = (unsigned short)(__float_as_uint(p) >> 16);
            }
        }
        __builtin_amdgcn_wave_barrier();  // order Pb writes vs reads (wave-local LDS, in-order DS)

        // A fragment of P (16x32 bf16): 2 x b128
        v16bf aP = loadA(&Pb[wave][r][0], 0);
        // B fragments of V via V^T: per lane keys {8h..+7} and {16+8h..+7} at fixed dim: 2 x b128
        #pragma unroll
        for (int nt = 0; nt < 4; ++nt) {
            const v8bf* vp = (const v8bf*)&Vt16[nt * 16 + r][8 * halfId];
            v16bf bV = __builtin_shufflevector(vp[0], vp[2],
                                               0,1,2,3,4,5,6,7,8,9,10,11,12,13,14,15);
            acc[nt] = __builtin_amdgcn_wmma_f32_16x16x32_bf16(false, aP, false, bV,
                                                              (short)0, acc[nt], false, false);
        }
        wait_async0();
        __syncthreads();
        fb ^= 1;
    }

    // ---- write context: C layout (M = i + 8*halfId, N = nt*16 + r) ----
    #pragma unroll
    for (int nt = 0; nt < 4; ++nt) {
        #pragma unroll
        for (int i = 0; i < 8; ++i) {
            int mrow = q0 + i + 8 * halfId;
            Cp[(size_t)mrow * D_DIM + nt * 16 + r] = acc[nt][i];
        }
    }
}

extern "C" void kernel_launch(void* const* d_in, const int* in_sizes, int n_in,
                              void* d_out, int out_size, void* d_ws, size_t ws_size,
                              hipStream_t stream) {
    (void)in_sizes; (void)n_in; (void)out_size; (void)d_ws; (void)ws_size;
    const float*         Q = (const float*)d_in[0];
    const float*         K = (const float*)d_in[1];
    const float*         V = (const float*)d_in[2];
    const unsigned char* M = (const unsigned char*)d_in[3];   // jnp.bool_ -> 1 byte/elem

    float* ctx  = (float*)d_out;                              // [2,16,2048,64]
    float* prob = ctx + (size_t)2 * 16 * 2048 * 64;           // [2,16,2048,2048]

    dim3 grid(S_LEN / 64, 2 * 16);   // 32 x 32 blocks, 128 thr (4 wave32) each
    sdpa_fused_kernel<<<grid, 128, 0, stream>>>(Q, K, V, M, ctx, prob);
}
